// NoisyQuantizedLinear_16475494548100
// MI455X (gfx1250) — matrix operand order repaired and checked by reading, not compiled, not run on hardware
//
#include <hip/hip_runtime.h>
#include <hip/hip_bf16.h>

// Problem constants (from reference)
#define BATCH   4096
#define IN_F    4096
#define OUT_F   11008
#define NGROUP  32          // IN_F / 128
#define ALPHA   0.05f

// GEMM tiling
#define BM 128
#define BN 128
#define BK 64
#define LDK (BK + 8)        // padded LDS stride (halfwords): 144B rows -> conflict-free

typedef __attribute__((ext_vector_type(16))) __bf16 bf16x16;
typedef __attribute__((ext_vector_type(8)))  float  floatx8;
typedef unsigned int u32x4 __attribute__((ext_vector_type(4)));
typedef int          i32x8 __attribute__((ext_vector_type(8)));
typedef int          i32x4 __attribute__((ext_vector_type(4)));

#if __has_builtin(__builtin_amdgcn_tensor_load_to_lds)
#define HAVE_TDM 1
#else
#define HAVE_TDM 0
#endif

union FragU { uint4 u[2]; bf16x16 v; };

__device__ __forceinline__ unsigned int f2bf(float f) {
    unsigned int u = __builtin_bit_cast(unsigned int, f);
    return (u + 0x7FFFu + ((u >> 16) & 1u)) >> 16;   // RNE f32->bf16
}

__device__ __forceinline__ unsigned int pack_bf16(float lo, float hi) {
#if __has_builtin(__builtin_amdgcn_cvt_pk_bf16_f32)
    typedef __bf16 bf16x2 __attribute__((ext_vector_type(2)));
    bf16x2 p = __builtin_amdgcn_cvt_pk_bf16_f32(lo, hi);
    return __builtin_bit_cast(unsigned int, p);
#else
    return f2bf(lo) | (f2bf(hi) << 16);
#endif
}

// ---------------------------------------------------------------------------
// Pass 1: dequant + noise -> bf16 w_noisy  (done ONCE: kills 32x repeated VALU)
// ---------------------------------------------------------------------------
__global__ __launch_bounds__(256)
void dequant_w_kernel(const int* __restrict__ qweight, const float* __restrict__ scales,
                      const float* __restrict__ zeros, const float* __restrict__ noise,
                      unsigned short* __restrict__ wbf)
{
    const size_t base = ((size_t)blockIdx.x * 256 + threadIdx.x) * 8;
    const int row = (int)(base >> 12);        // / IN_F
    const int col = (int)(base & (IN_F - 1));
    const int g   = col >> 7;                 // 8 elems never straddle a 128-group
    const float sc = scales[(size_t)row * NGROUP + g];
    const float zs = zeros [(size_t)row * NGROUP + g] * sc;

    const int4   q0 = *(const int4*)  (qweight + base);
    const int4   q1 = *(const int4*)  (qweight + base + 4);
    const float4 n0 = *(const float4*)(noise   + base);
    const float4 n1 = *(const float4*)(noise   + base + 4);

    float w[8];
    w[0] = fmaf((float)q0.x, sc, -zs); w[1] = fmaf((float)q0.y, sc, -zs);
    w[2] = fmaf((float)q0.z, sc, -zs); w[3] = fmaf((float)q0.w, sc, -zs);
    w[4] = fmaf((float)q1.x, sc, -zs); w[5] = fmaf((float)q1.y, sc, -zs);
    w[6] = fmaf((float)q1.z, sc, -zs); w[7] = fmaf((float)q1.w, sc, -zs);
    const float nn[8] = {n0.x, n0.y, n0.z, n0.w, n1.x, n1.y, n1.z, n1.w};
#pragma unroll
    for (int i = 0; i < 8; ++i) w[i] = fmaf(ALPHA * nn[i], fabsf(w[i]), w[i]);

    uint4 p;
    p.x = pack_bf16(w[0], w[1]); p.y = pack_bf16(w[2], w[3]);
    p.z = pack_bf16(w[4], w[5]); p.w = pack_bf16(w[6], w[7]);
    *(uint4*)(wbf + base) = p;
}

// ---------------------------------------------------------------------------
// Pass 2: x (f32) -> bf16  (done ONCE instead of once per N-block)
// ---------------------------------------------------------------------------
__global__ __launch_bounds__(256)
void cvt_x_kernel(const float* __restrict__ x, unsigned short* __restrict__ xbf)
{
    const size_t base = ((size_t)blockIdx.x * 256 + threadIdx.x) * 8;
    const float4 a = *(const float4*)(x + base);
    const float4 b = *(const float4*)(x + base + 4);
    uint4 p;
    p.x = pack_bf16(a.x, a.y); p.y = pack_bf16(a.z, a.w);
    p.z = pack_bf16(b.x, b.y); p.w = pack_bf16(b.z, b.w);
    *(uint4*)(xbf + base) = p;
}

// ---------------------------------------------------------------------------
// Pass 3: bf16 WMMA GEMM; tiles fetched by Tensor Data Mover (double-buffered)
// ---------------------------------------------------------------------------
#if HAVE_TDM
// 2D tile load: tile_dim0=BK cols (contiguous), tile_dim1=128 rows, element=2B,
// row stride = IN_F elements; TDM pads LDS dest 16B per 128B row -> stride LDK.
__device__ __forceinline__ void tdm_load_tile(unsigned lds_off,
                                              const unsigned short* gptr,
                                              unsigned tensor_rows)
{
    const unsigned long long ga = (unsigned long long)(uintptr_t)gptr;
    u32x4 g0;
    g0[0] = 1u;                                             // count=1, user D#
    g0[1] = lds_off;                                        // LDS byte address
    g0[2] = (unsigned)(ga & 0xffffffffu);                   // global_addr[31:0]
    g0[3] = (unsigned)((ga >> 32) & 0x1ffffffu) | (2u << 30); // addr[56:32] | type=2
    i32x8 g1;
    g1[0] = (1 << 16) | (1 << 20) | (4 << 22) | (3 << 25);  // 2B elems, pad: +4dw / 32dw
    g1[1] = (int)((IN_F & 0xffff) << 16);                   // tensor_dim0 lo16
    g1[2] = (int)(((IN_F >> 16) & 0xffff) | ((tensor_rows & 0xffff) << 16)); // dim0 hi / dim1 lo
    g1[3] = (int)(((tensor_rows >> 16) & 0xffff) | (BK << 16));              // dim1 hi / tile_dim0
    g1[4] = BM;                                             // tile_dim1=128, tile_dim2=0
    g1[5] = IN_F;                                           // tensor_dim0_stride lo32
    g1[6] = 0;                                              // stride hi / dim1_stride lo
    g1[7] = 0;
    const i32x4 z4 = {0, 0, 0, 0};                          // groups 2/3 unused (2D tile)
    const i32x8 z8 = {0, 0, 0, 0, 0, 0, 0, 0};
    __builtin_amdgcn_tensor_load_to_lds(g0, g1, z4, z4, z8, 0);  // 6-arg toolchain form
}
#define NBUF 2
#else
#define NBUF 1
#endif

__global__ __launch_bounds__(256)
void gemm_bf16_kernel(const unsigned short* __restrict__ xbf,
                      const unsigned short* __restrict__ wbf,
                      const float* __restrict__ bias,
                      float* __restrict__ out)
{
    __shared__ __align__(16) unsigned short As[NBUF][BM * LDK];
    __shared__ __align__(16) unsigned short Bs[NBUF][BN * LDK];

    const int tid  = threadIdx.x;
    const int lane = tid & 31;
    const int wave = tid >> 5;
    const int wm   = wave >> 2;          // 0..1 : 64-row M slice
    const int wn   = wave & 3;           // 0..3 : 32-col N slice

    const int mBase = blockIdx.x * BM;   // M fastest -> weight strip L2 reuse, x L2-resident
    const int nBase = blockIdx.y * BN;

    floatx8 acc[4][2];
#pragma unroll
    for (int i = 0; i < 4; ++i)
#pragma unroll
        for (int j = 0; j < 2; ++j)
#pragma unroll
            for (int r = 0; r < 8; ++r) acc[i][j][r] = 0.0f;

    constexpr int NK = IN_F / BK;        // 64 K-tiles

#if HAVE_TDM
    const unsigned ldsA0 = (unsigned)(uintptr_t)&As[0][0];
    const unsigned ldsA1 = (unsigned)(uintptr_t)&As[1][0];
    const unsigned ldsB0 = (unsigned)(uintptr_t)&Bs[0][0];
    const unsigned ldsB1 = (unsigned)(uintptr_t)&Bs[1][0];
    if (wave == 0) {                     // prologue: DMA tile 0
        tdm_load_tile(ldsA0, xbf + (size_t)mBase * IN_F, BATCH);
        tdm_load_tile(ldsB0, wbf + (size_t)nBase * IN_F, OUT_F);
    }
#endif

    for (int it = 0; it < NK; ++it) {
        const int buf = it & (NBUF - 1);
#if HAVE_TDM
        if (wave == 0) __builtin_amdgcn_s_wait_tensorcnt(0);   // tile it landed
        __syncthreads();                 // tile visible; prev compute done everywhere
        if (wave == 0 && it + 1 < NK) {  // overlap: DMA tile it+1 into other buffer
            const int kn = (it + 1) * BK;
            tdm_load_tile((it + 1) & 1 ? ldsA1 : ldsA0,
                          xbf + (size_t)mBase * IN_F + kn, BATCH);
            tdm_load_tile((it + 1) & 1 ? ldsB1 : ldsB0,
                          wbf + (size_t)nBase * IN_F + kn, OUT_F);
        }
#else
        // fallback staging: 256 threads copy two 128x64 bf16 tiles (4 uint4 each)
        const int kt  = it * BK;
        uint4 ra[4], rb[4];
#pragma unroll
        for (int s = 0; s < 4; ++s) {
            const int t4  = tid + s * 256;
            const int row = t4 >> 3;
            const int c8  = (t4 & 7) << 3;
            ra[s] = *(const uint4*)(xbf + (size_t)(mBase + row) * IN_F + kt + c8);
            rb[s] = *(const uint4*)(wbf + (size_t)(nBase + row) * IN_F + kt + c8);
        }
        __syncthreads();
#pragma unroll
        for (int s = 0; s < 4; ++s) {
            const int t4  = tid + s * 256;
            const int row = t4 >> 3;
            const int c8  = (t4 & 7) << 3;
            *(uint4*)&As[0][row * LDK + c8] = ra[s];
            *(uint4*)&Bs[0][row * LDK + c8] = rb[s];
        }
        __syncthreads();
#endif

#pragma unroll
        for (int ks = 0; ks < BK; ks += 32) {
            FragU a[4], b[2];
            // A 16x32 bf16: lanes 0-15 hold K {0..7,16..23}, lanes 16-31 {8..15,24..31}
            const int aK = ks + ((lane & 16) ? 8 : 0);
#pragma unroll
            for (int i = 0; i < 4; ++i) {
                const unsigned short* p =
                    &As[buf][(wm * 64 + i * 16 + (lane & 15)) * LDK + aK];
                a[i].u[0] = *(const uint4*)p;
                a[i].u[1] = *(const uint4*)(p + 16);
            }
            // B 32x16 bf16: lanes 0-15 hold K {0..7,16..23}, lanes 16-31 {8..15,24..31}
            const int bK = ks + ((lane & 16) ? 8 : 0);
#pragma unroll
            for (int j = 0; j < 2; ++j) {
                const unsigned short* p =
                    &Bs[buf][(wn * 32 + j * 16 + (lane & 15)) * LDK + bK];
                b[j].u[0] = *(const uint4*)p;
                b[j].u[1] = *(const uint4*)(p + 16);
            }
#pragma unroll
            for (int i = 0; i < 4; ++i)
#pragma unroll
                for (int j = 0; j < 2; ++j)
                    acc[i][j] = __builtin_amdgcn_wmma_f32_16x16x32_bf16(
                        false, a[i].v, false, b[j].v,
                        (short)0, acc[i][j], false, false);
        }
    }

    // Epilogue: C/D layout — VGPR r: lanes 0-15 -> row r, lanes 16-31 -> row r+8
    const int colLane = lane & 15;
    const int rowHalf = (lane >> 4) << 3;
#pragma unroll
    for (int j = 0; j < 2; ++j) {
        const int col = nBase + wn * 32 + j * 16 + colLane;
        const float bb = bias[col];
#pragma unroll
        for (int i = 0; i < 4; ++i) {
            const int rbase = mBase + wm * 64 + i * 16 + rowHalf;
#pragma unroll
            for (int r = 0; r < 8; ++r)
                out[(size_t)(rbase + r) * OUT_F + col] = acc[i][j][r] + bb;
        }
    }
}

// ---------------------------------------------------------------------------
// Fallback: single fused kernel (used only if d_ws is too small) — round-1 code
// ---------------------------------------------------------------------------
__global__ __launch_bounds__(256)
void nql_fused_kernel(const float* __restrict__ x, const int* __restrict__ qweight,
                      const float* __restrict__ scales, const float* __restrict__ zeros,
                      const float* __restrict__ bias, const float* __restrict__ noise,
                      float* __restrict__ out)
{
    __shared__ __align__(16) unsigned short As[BM * LDK];
    __shared__ __align__(16) unsigned short Bs[BN * LDK];

    const int tid  = threadIdx.x;
    const int lane = tid & 31;
    const int wave = tid >> 5;
    const int wm   = wave >> 2;
    const int wn   = wave & 3;
    const int mBase = blockIdx.x * BM;
    const int nBase = blockIdx.y * BN;

    floatx8 acc[4][2];
#pragma unroll
    for (int i = 0; i < 4; ++i)
#pragma unroll
        for (int j = 0; j < 2; ++j)
#pragma unroll
            for (int r = 0; r < 8; ++r) acc[i][j][r] = 0.0f;

    const int srow = tid >> 4;
    const int scol = (tid & 15) << 2;

    for (int kt = 0; kt < IN_F; kt += BK) {
        const int g = kt >> 7;
        float4 ax[8]; int4 qv[8]; float4 nv[8]; float sc8[8], zp8[8];
#pragma unroll
        for (int s = 0; s < 8; ++s) {
            const int row = srow + s * 16;
            ax[s]  = *(const float4*)(x       + (size_t)(mBase + row) * IN_F + kt + scol);
            qv[s]  = *(const int4*)  (qweight + (size_t)(nBase + row) * IN_F + kt + scol);
            nv[s]  = *(const float4*)(noise   + (size_t)(nBase + row) * IN_F + kt + scol);
            sc8[s] = scales[(size_t)(nBase + row) * NGROUP + g];
            zp8[s] = zeros [(size_t)(nBase + row) * NGROUP + g];
        }
        __syncthreads();
#pragma unroll
        for (int s = 0; s < 8; ++s) {
            const int row = srow + s * 16;
            uint2 pa;
            pa.x = pack_bf16(ax[s].x, ax[s].y);
            pa.y = pack_bf16(ax[s].z, ax[s].w);
            *(uint2*)&As[row * LDK + scol] = pa;
            const float sc = sc8[s], zs = zp8[s] * sc;
            float w0 = fmaf((float)qv[s].x, sc, -zs);
            float w1 = fmaf((float)qv[s].y, sc, -zs);
            float w2 = fmaf((float)qv[s].z, sc, -zs);
            float w3 = fmaf((float)qv[s].w, sc, -zs);
            w0 = fmaf(ALPHA * nv[s].x, fabsf(w0), w0);
            w1 = fmaf(ALPHA * nv[s].y, fabsf(w1), w1);
            w2 = fmaf(ALPHA * nv[s].z, fabsf(w2), w2);
            w3 = fmaf(ALPHA * nv[s].w, fabsf(w3), w3);
            uint2 pb;
            pb.x = pack_bf16(w0, w1);
            pb.y = pack_bf16(w2, w3);
            *(uint2*)&Bs[row * LDK + scol] = pb;
        }
        __syncthreads();
#pragma unroll
        for (int ks = 0; ks < BK; ks += 32) {
            FragU a[4], b[2];
            const int aK = ks + ((lane & 16) ? 8 : 0);
#pragma unroll
            for (int i = 0; i < 4; ++i) {
                const unsigned short* p = &As[(wm * 64 + i * 16 + (lane & 15)) * LDK + aK];
                a[i].u[0] = *(const uint4*)p;
                a[i].u[1] = *(const uint4*)(p + 16);
            }
            const int bK = ks + ((lane & 16) ? 8 : 0);
#pragma unroll
            for (int j = 0; j < 2; ++j) {
                const unsigned short* p = &Bs[(wn * 32 + j * 16 + (lane & 15)) * LDK + bK];
                b[j].u[0] = *(const uint4*)p;
                b[j].u[1] = *(const uint4*)(p + 16);
            }
#pragma unroll
            for (int i = 0; i < 4; ++i)
#pragma unroll
                for (int j = 0; j < 2; ++j)
                    acc[i][j] = __builtin_amdgcn_wmma_f32_16x16x32_bf16(
                        false, a[i].v, false, b[j].v, (short)0, acc[i][j], false, false);
        }
    }

    const int colLane = lane & 15;
    const int rowHalf = (lane >> 4) << 3;
#pragma unroll
    for (int j = 0; j < 2; ++j) {
        const int col = nBase + wn * 32 + j * 16 + colLane;
        const float bb = bias[col];
#pragma unroll
        for (int i = 0; i < 4; ++i) {
            const int rbase = mBase + wm * 64 + i * 16 + rowHalf;
#pragma unroll
            for (int r = 0; r < 8; ++r)
                out[(size_t)(rbase + r) * OUT_F + col] = acc[i][j][r] + bb;
        }
    }
}

extern "C" void kernel_launch(void* const* d_in, const int* in_sizes, int n_in,
                              void* d_out, int out_size, void* d_ws, size_t ws_size,
                              hipStream_t stream) {
    const float* x       = (const float*)d_in[0];
    const int*   qweight = (const int*)  d_in[1];
    const float* scales  = (const float*)d_in[2];
    const float* zeros   = (const float*)d_in[3];
    const float* bias    = (const float*)d_in[4];
    const float* noise   = (const float*)d_in[5];
    float*       out     = (float*)d_out;

    const size_t WBF_BYTES = (size_t)OUT_F * IN_F * 2;   // 90.2 MB
    const size_t XBF_BYTES = (size_t)BATCH * IN_F * 2;   // 33.6 MB

    if (ws_size >= WBF_BYTES + XBF_BYTES) {
        unsigned short* wbf = (unsigned short*)d_ws;
        unsigned short* xbf = (unsigned short*)((char*)d_ws + WBF_BYTES);
        dequant_w_kernel<<<(int)(((size_t)OUT_F * IN_F) / (256 * 8)), 256, 0, stream>>>(
            qweight, scales, zeros, noise, wbf);
        cvt_x_kernel<<<(int)(((size_t)BATCH * IN_F) / (256 * 8)), 256, 0, stream>>>(x, xbf);
        gemm_bf16_kernel<<<dim3(BATCH / BM, OUT_F / BN), 256, 0, stream>>>(xbf, wbf, bias, out);
    } else {
        nql_fused_kernel<<<dim3(BATCH / BM, OUT_F / BN), 256, 0, stream>>>(
            x, qweight, scales, zeros, bias, noise, out);
    }
}
